// MoDLayer_57174604645051
// MI455X (gfx1250) — compile-verified, matrix operand order checked by reading, not run on hardware
//
#include <hip/hip_runtime.h>
#include <math.h>

// ---------------- types / helpers ----------------
typedef __attribute__((ext_vector_type(16))) __bf16 v16bf;
typedef __attribute__((ext_vector_type(8)))  float  v8f;
typedef __attribute__((ext_vector_type(4)))  unsigned u32x4;
typedef __attribute__((ext_vector_type(4)))  float  f32x4;

union BFrag { v16bf v; u32x4 q[2]; };
union FAcc  { v8f  v; float    f[8]; };

// CDNA5 async global->LDS path (ASYNCcnt), compile-safe guarded
#if defined(__has_builtin)
# if __has_builtin(__builtin_amdgcn_global_load_async_to_lds_b128)
#  define HAVE_ASYNC_LDS 1
# endif
#endif

// builtin signature (from hipcc diagnostic): (global int4*, lds int4*, imm offset, imm cpol)
typedef int v4i __attribute__((__vector_size__(16)));
typedef __attribute__((address_space(1))) v4i gv4i;
typedef __attribute__((address_space(3))) v4i lv4i;

__device__ __forceinline__ void async_ld_b128(const void* g, void* l) {
#ifdef HAVE_ASYNC_LDS
    __builtin_amdgcn_global_load_async_to_lds_b128((gv4i*)g, (lv4i*)l, 0, 0);
#else
    (void)g; (void)l;
#endif
}
__device__ __forceinline__ void wait_async() {
#ifdef HAVE_ASYNC_LDS
# if __has_builtin(__builtin_amdgcn_s_wait_asynccnt)
    __builtin_amdgcn_s_wait_asynccnt(0);
# else
    asm volatile("s_wait_asynccnt 0x0" ::: "memory");
# endif
#endif
}

__device__ __forceinline__ unsigned short f2bf(float x) {
    union { float f; unsigned u; } c; c.f = x;
    unsigned u = c.u;
    unsigned r = (u >> 16) & 1u;
    u += 0x7fffu + r;
    return (unsigned short)(u >> 16);
}

// one 16x32 bf16 A/B fragment from an LDS row (row base 16B-aligned):
// per lane, bytes [16*hs, +16) and [32+16*hs, +16) of the 64-byte K-chunk row.
__device__ __forceinline__ void load_frag(BFrag& f, const unsigned short* base, int hs) {
    const u32x4* p = (const u32x4*)base;
    f.q[0] = p[hs];
    f.q[1] = p[2 + hs];
}

__device__ __forceinline__ float block_sum256(float v) {
    #pragma unroll
    for (int m = 16; m > 0; m >>= 1) v += __shfl_xor(v, m, 32);
    __shared__ float red[8];
    int lane = threadIdx.x & 31, w = threadIdx.x >> 5;
    if (lane == 0) red[w] = v;
    __syncthreads();
    float tot = 0.f;
    #pragma unroll
    for (int i = 0; i < 8; i++) tot += red[i];
    return tot;
}

// ---------------- 1) router scores ----------------
__global__ __launch_bounds__(256) void router_kernel(
    const float* __restrict__ hidden, const float* __restrict__ wr,
    float* __restrict__ scores)
{
    int tok  = (blockIdx.x * 256 + threadIdx.x) >> 5;
    int lane = threadIdx.x & 31;
    if (tok >= 16384) return;
    const float* h = hidden + (size_t)tok * 2048;
    float acc = 0.f;
    for (int i = lane; i < 2048; i += 32) acc += h[i] * wr[i];
    #pragma unroll
    for (int m = 16; m > 0; m >>= 1) acc += __shfl_xor(acc, m, 32);
    if (lane == 0) scores[tok] = acc;
}

// ---------------- 2) per-batch top-512 of 4096 (bitonic in LDS) ----------------
__global__ __launch_bounds__(1024) void topk_kernel(
    const float* __restrict__ scores, int* __restrict__ idx_out)
{
    __shared__ float sv[4096];
    __shared__ int   si[4096];
    int b = blockIdx.x, t = threadIdx.x;
    for (int i = t; i < 4096; i += 1024) { sv[i] = scores[b * 4096 + i]; si[i] = i; }
    __syncthreads();
    for (int k = 2; k <= 4096; k <<= 1) {
        for (int j = k >> 1; j > 0; j >>= 1) {
            for (int i = t; i < 4096; i += 1024) {
                int ixj = i ^ j;
                if (ixj > i) {
                    bool dir = (i & k) != 0;
                    bool sw  = sv[i] < sv[ixj];
                    if (sw != dir) {
                        float tv = sv[i]; sv[i] = sv[ixj]; sv[ixj] = tv;
                        int   ti = si[i]; si[i] = si[ixj]; si[ixj] = ti;
                    }
                }
            }
            __syncthreads();
        }
    }
    for (int k = 2; k <= 512; k <<= 1) {
        for (int j = k >> 1; j > 0; j >>= 1) {
            if (t < 512) {
                int i = t, ixj = i ^ j;
                if (ixj > i) {
                    bool dir = (i & k) == 0;
                    bool sw  = si[i] > si[ixj];
                    if (sw == dir) { int ti = si[i]; si[i] = si[ixj]; si[ixj] = ti; }
                }
            }
            __syncthreads();
        }
    }
    for (int i = t; i < 512; i += 1024) idx_out[b * 512 + i] = si[i];
}

// ---------------- 3) gather + ln1 ----------------
__global__ __launch_bounds__(256) void gather_ln1(
    const float* __restrict__ hidden, const float* __restrict__ scores,
    const int* __restrict__ idx, const float* __restrict__ ln1w,
    float* __restrict__ xsel, unsigned short* __restrict__ hb,
    float* __restrict__ gate, int* __restrict__ pos)
{
    int s = blockIdx.x;
    int b = s >> 9, j = s & 511;
    int p = idx[b * 512 + j];
    const float* src = hidden + ((size_t)b * 4096 + p) * 2048;
    float loc[8], ss = 0.f;
    #pragma unroll
    for (int i = 0; i < 8; i++) {
        float v = src[threadIdx.x + i * 256];
        loc[i] = v; ss += v * v;
    }
    float tot = block_sum256(ss);
    float scale = rsqrtf(tot / 2048.f + 1e-6f);
    #pragma unroll
    for (int i = 0; i < 8; i++) {
        int c = threadIdx.x + i * 256;
        xsel[(size_t)s * 2048 + c] = loc[i];
        hb[(size_t)s * 2048 + c]   = f2bf(loc[i] * scale * ln1w[c]);
    }
    if (threadIdx.x == 0) {
        pos[s]  = p;
        gate[s] = 1.f / (1.f + expf(-scores[b * 4096 + p]));
    }
}

// ---------------- bf16 WMMA GEMM, double-buffered LDS, async A staging ----------------
// C[M,N] = A[M,K](bf16) @ Bw[K,N](f32->bf16 on stage); block tile 128x128, K-step 32.
// MODE 0: Cf = acc + bias (bias nullable)    MODE 1: Cb = bf16(acc + bias)
// MODE 3: Cb = bf16(silu(aux)*acc)           MODE 4: gated residual scatter into outFull
#define TSTR 40   // 80B LDS rows: 16B aligned, conflict-free b128 fragment loads
template<int MODE>
__global__ __launch_bounds__(256) void gemm_wmma(
    const unsigned short* __restrict__ A, const float* __restrict__ Bw,
    const float* __restrict__ bias,
    float* __restrict__ Cf, unsigned short* __restrict__ Cb,
    const float* __restrict__ aux, const float* __restrict__ aux2,
    const float* __restrict__ gatep, const int* __restrict__ posp,
    float* __restrict__ outFull,
    int M, int N, int K)
{
    __shared__ unsigned short As[2][128 * TSTR];
    __shared__ unsigned short Bs[2][128 * TSTR];   // transposed: [n][k]
    int tid = threadIdx.x, lane = tid & 31, w = tid >> 5;
    int m0 = blockIdx.y * 128, n0 = blockIdx.x * 128;
    int r = lane & 15, hs = lane >> 4;

    FAcc acc[8];
    #pragma unroll
    for (int tn = 0; tn < 8; tn++)
        #pragma unroll
        for (int i = 0; i < 8; i++) acc[tn].f[i] = 0.f;

    int arow = tid >> 1, ahalf = tid & 1;        // A: 16 bf16 (2 x b128) per thread
    int bk = tid >> 3,  bn = (tid & 7) * 16;     // B: 16 f32 per thread

    f32x4 bReg[4];
    auto stageB = [&](int kc) {
        const f32x4* gb = (const f32x4*)(Bw + (size_t)(kc + bk) * N + n0 + bn);
        #pragma unroll
        for (int i = 0; i < 4; i++) bReg[i] = gb[i];
        if (kc + 32 < K)
            __builtin_prefetch(Bw + (size_t)(kc + 32 + bk) * N + n0 + bn, 0, 0);
    };
    auto commitB = [&](int buf) {
        const float* fp = (const float*)bReg;
        #pragma unroll
        for (int i = 0; i < 16; i++) Bs[buf][(bn + i) * TSTR + bk] = f2bf(fp[i]);
    };

#ifdef HAVE_ASYNC_LDS
    auto stageA = [&](int kc, int buf) {   // async global -> LDS, no VGPR staging
        const unsigned short* g = A + (size_t)(m0 + arow) * K + kc + ahalf * 16;
        unsigned short* l = &As[buf][arow * TSTR + ahalf * 16];
        async_ld_b128(g, l);
        async_ld_b128(g + 8, l + 8);
    };
    stageA(0, 0);
#else
    u32x4 aReg[2];
    auto stageA = [&](int kc) {
        const u32x4* ga = (const u32x4*)(A + (size_t)(m0 + arow) * K + kc + ahalf * 16);
        aReg[0] = ga[0]; aReg[1] = ga[1];
    };
    auto commitA = [&](int buf) {
        u32x4* la = (u32x4*)&As[buf][arow * TSTR + ahalf * 16];
        la[0] = aReg[0]; la[1] = aReg[1];
    };
    stageA(0);
    commitA(0);
#endif
    stageB(0);
    commitB(0);
    wait_async();
    __syncthreads();

    int cur = 0;
    for (int kc = 0; kc < K; kc += 32) {
        bool more = (kc + 32) < K;
        if (more) {
#ifdef HAVE_ASYNC_LDS
            stageA(kc + 32, cur ^ 1);
#else
            stageA(kc + 32);
#endif
            stageB(kc + 32);
        }

        BFrag af;
        load_frag(af, &As[cur][(w * 16 + r) * TSTR], hs);
        BFrag bf[8];
        #pragma unroll
        for (int tn = 0; tn < 8; tn++)
            load_frag(bf[tn], &Bs[cur][(tn * 16 + r) * TSTR], hs);
        #pragma unroll
        for (int tn = 0; tn < 8; tn++)
            acc[tn].v = __builtin_amdgcn_wmma_f32_16x16x32_bf16(
                false, af.v, false, bf[tn].v, (short)0, acc[tn].v, false, false);

        if (more) {
            commitB(cur ^ 1);
#ifndef HAVE_ASYNC_LDS
            commitA(cur ^ 1);
#endif
        }
        wait_async();
        __syncthreads();
        cur ^= 1;
    }

    // epilogue
    #pragma unroll
    for (int tn = 0; tn < 8; tn++) {
        int gn = n0 + tn * 16 + r;
        float bval = 0.f;
        if (MODE <= 1 && bias) bval = bias[gn];
        #pragma unroll
        for (int v = 0; v < 8; v++) {
            int gm = m0 + w * 16 + v + 8 * hs;
            float val = acc[tn].f[v];
            size_t ij = (size_t)gm * N + gn;
            if (MODE == 0) {
                Cf[ij] = val + bval;
            } else if (MODE == 1) {
                Cb[ij] = f2bf(val + bval);
            } else if (MODE == 3) {
                float g = aux[ij];
                Cb[ij] = f2bf((g / (1.f + expf(-g))) * val);
            } else if (MODE == 4) {
                float proc = aux[ij] + val;
                float gt   = gatep[gm];
                float xs   = aux2[ij];
                size_t orow = (size_t)(gm >> 9) * 4096 + (size_t)posp[gm];
                outFull[orow * 2048 + gn] = gt * proc + (1.f - gt) * xs;
            }
        }
    }
}

// ---------------- RoPE ----------------
__global__ __launch_bounds__(256) void rope_kernel(
    const float* __restrict__ qf, const float* __restrict__ kf,
    const int* __restrict__ pos,
    unsigned short* __restrict__ qb, unsigned short* __restrict__ kb)
{
    int s = blockIdx.x;
    float p = (float)pos[s];
    for (int wi = threadIdx.x; wi < 1536; wi += 256) {
        int head = wi >> 6, d = wi & 63;
        float inv = expf(-((float)(2 * d) / 128.f) * 9.210340372f);
        float ang = p * inv;
        float c = cosf(ang), sn = sinf(ang);
        if (head < 16) {
            const float* qr = qf + (size_t)s * 2048 + head * 128;
            float a = qr[d], b2 = qr[d + 64];
            qb[(size_t)s * 2048 + head * 128 + d]      = f2bf(a * c - b2 * sn);
            qb[(size_t)s * 2048 + head * 128 + d + 64] = f2bf(b2 * c + a * sn);
        } else {
            int kh = head - 16;
            const float* kr = kf + (size_t)s * 1024 + kh * 128;
            float a = kr[d], b2 = kr[d + 64];
            kb[(size_t)s * 1024 + kh * 128 + d]      = f2bf(a * c - b2 * sn);
            kb[(size_t)s * 1024 + kh * 128 + d + 64] = f2bf(b2 * c + a * sn);
        }
    }
}

// ---------------- flash attention (causal, GQA), bf16 WMMA ----------------
// grid (S/128, H). Dynamic LDS: Qs,Ks,VsT,Ps each 128 x 136 bf16 (~136 KB).
#define QSTR 136  // 272B rows: 16B-aligned, conflict-free b128 fragment loads
__global__ __launch_bounds__(256) void flash_attn(
    const unsigned short* __restrict__ qb, const unsigned short* __restrict__ kb,
    const unsigned short* __restrict__ vb, unsigned short* __restrict__ ob)
{
    extern __shared__ unsigned short smem[];
    unsigned short* Qs  = smem;
    unsigned short* Ks  = Qs  + 128 * QSTR;
    unsigned short* VsT = Ks  + 128 * QSTR;   // [d][key]
    unsigned short* Ps  = VsT + 128 * QSTR;   // [128 rows][key]

    int tid = threadIdx.x, lane = tid & 31, w = tid >> 5;
    int qt = blockIdx.x, h = blockIdx.y, kvh = h >> 1;
    int q0 = qt * 128;
    int r = lane & 15, hs = lane >> 4;

    int lrow = tid >> 1, lseg = (tid & 1) * 64;
    {   // Q tile: 8 x b128 per thread (async if available)
        const unsigned short* gq = qb + (size_t)(q0 + lrow) * 2048 + h * 128 + lseg;
        unsigned short* lq = &Qs[lrow * QSTR + lseg];
#ifdef HAVE_ASYNC_LDS
        #pragma unroll
        for (int i = 0; i < 8; i++) async_ld_b128(gq + i * 8, lq + i * 8);
#else
        const u32x4* g = (const u32x4*)gq;
        u32x4* l = (u32x4*)lq;
        #pragma unroll
        for (int i = 0; i < 8; i++) l[i] = g[i];
#endif
    }

    FAcc oacc[8];
    #pragma unroll
    for (int dt = 0; dt < 8; dt++)
        #pragma unroll
        for (int i = 0; i < 8; i++) oacc[dt].f[i] = 0.f;
    float mrow[8], lsum[8];
    #pragma unroll
    for (int v = 0; v < 8; v++) { mrow[v] = -1e30f; lsum[v] = 0.f; }

    for (int kt = 0; kt <= qt; kt++) {
        __syncthreads();   // all waves done reading previous tiles
        int krow = kt * 128 + lrow;
        {   // K tile [key][d]
            const unsigned short* gk = kb + (size_t)krow * 1024 + kvh * 128 + lseg;
            unsigned short* lk = &Ks[lrow * QSTR + lseg];
#ifdef HAVE_ASYNC_LDS
            #pragma unroll
            for (int i = 0; i < 8; i++) async_ld_b128(gk + i * 8, lk + i * 8);
#else
            const u32x4* g = (const u32x4*)gk;
            u32x4* l = (u32x4*)lk;
            #pragma unroll
            for (int i = 0; i < 8; i++) l[i] = g[i];
#endif
        }
        {   // V tile transposed -> [d][key] (needs per-element transpose)
            const unsigned short* gv = vb + (size_t)krow * 1024 + kvh * 128 + lseg;
            #pragma unroll
            for (int i = 0; i < 64; i++) VsT[(lseg + i) * QSTR + lrow] = gv[i];
        }
        wait_async();
        __syncthreads();

        // S = Q K^T  (batched fragment loads, groups of 4)
        FAcc sacc[8];
        #pragma unroll
        for (int tn = 0; tn < 8; tn++)
            #pragma unroll
            for (int i = 0; i < 8; i++) sacc[tn].f[i] = 0.f;
        #pragma unroll
        for (int kc = 0; kc < 128; kc += 32) {
            BFrag af;
            load_frag(af, &Qs[(w * 16 + r) * QSTR + kc], hs);
            #pragma unroll
            for (int g4 = 0; g4 < 2; g4++) {
                BFrag bf[4];
                #pragma unroll
                for (int i = 0; i < 4; i++)
                    load_frag(bf[i], &Ks[((g4 * 4 + i) * 16 + r) * QSTR + kc], hs);
                #pragma unroll
                for (int i = 0; i < 4; i++)
                    sacc[g4 * 4 + i].v = __builtin_amdgcn_wmma_f32_16x16x32_bf16(
                        false, af.v, false, bf[i].v, (short)0, sacc[g4 * 4 + i].v, false, false);
            }
        }

        const float scl = 0.08838834764831845f;  // 1/sqrt(128)
        #pragma unroll
        for (int v = 0; v < 8; v++) {
            int qrow = q0 + w * 16 + v + 8 * hs;
            float rmax = -1e30f;
            #pragma unroll
            for (int tn = 0; tn < 8; tn++) {
                float sv = sacc[tn].f[v] * scl;
                int key = kt * 128 + tn * 16 + r;
                if (key > qrow) sv = -1e30f;
                sacc[tn].f[v] = sv;
                rmax = fmaxf(rmax, sv);
            }
            #pragma unroll
            for (int m = 1; m < 16; m <<= 1) rmax = fmaxf(rmax, __shfl_xor(rmax, m, 32));
            float mnew = fmaxf(mrow[v], rmax);
            float so   = expf(mrow[v] - mnew);
            float psum = 0.f;
            #pragma unroll
            for (int tn = 0; tn < 8; tn++) {
                float pe = expf(sacc[tn].f[v] - mnew);
                sacc[tn].f[v] = pe;
                psum += pe;
            }
            #pragma unroll
            for (int m = 1; m < 16; m <<= 1) psum += __shfl_xor(psum, m, 32);
            lsum[v] = lsum[v] * so + psum;
            mrow[v] = mnew;
            #pragma unroll
            for (int dt = 0; dt < 8; dt++) oacc[dt].f[v] *= so;
            #pragma unroll
            for (int tn = 0; tn < 8; tn++)
                Ps[(w * 16 + v + 8 * hs) * QSTR + tn * 16 + r] = f2bf(sacc[tn].f[v]);
        }

        // O += P V
        #pragma unroll
        for (int kc = 0; kc < 128; kc += 32) {
            BFrag pf;
            load_frag(pf, &Ps[(w * 16 + r) * QSTR + kc], hs);
            #pragma unroll
            for (int g4 = 0; g4 < 2; g4++) {
                BFrag vf[4];
                #pragma unroll
                for (int i = 0; i < 4; i++)
                    load_frag(vf[i], &VsT[((g4 * 4 + i) * 16 + r) * QSTR + kc], hs);
                #pragma unroll
                for (int i = 0; i < 4; i++)
                    oacc[g4 * 4 + i].v = __builtin_amdgcn_wmma_f32_16x16x32_bf16(
                        false, pf.v, false, vf[i].v, (short)0, oacc[g4 * 4 + i].v, false, false);
            }
        }
    }

    #pragma unroll
    for (int v = 0; v < 8; v++) {
        int row = q0 + w * 16 + v + 8 * hs;
        float inv = 1.f / lsum[v];
        #pragma unroll
        for (int dt = 0; dt < 8; dt++)
            ob[(size_t)row * 2048 + h * 128 + dt * 16 + r] = f2bf(oacc[dt].f[v] * inv);
    }
}

// ---------------- residual + ln2 ----------------
__global__ __launch_bounds__(256) void resid_ln2(
    const float* __restrict__ xsel, const float* __restrict__ addv,
    const float* __restrict__ w2,
    float* __restrict__ x1, unsigned short* __restrict__ hb)
{
    int s = blockIdx.x;
    const float* xr = xsel + (size_t)s * 2048;
    const float* ar = addv + (size_t)s * 2048;
    float loc[8], ss = 0.f;
    #pragma unroll
    for (int i = 0; i < 8; i++) {
        int c = threadIdx.x + i * 256;
        float v = xr[c] + ar[c];
        loc[i] = v; ss += v * v;
    }
    float tot = block_sum256(ss);
    float scale = rsqrtf(tot / 2048.f + 1e-6f);
    #pragma unroll
    for (int i = 0; i < 8; i++) {
        int c = threadIdx.x + i * 256;
        x1[(size_t)s * 2048 + c] = loc[i];
        hb[(size_t)s * 2048 + c] = f2bf(loc[i] * scale * w2[c]);
    }
}

// ---------------- launch ----------------
extern "C" void kernel_launch(void* const* d_in, const int* in_sizes, int n_in,
                              void* d_out, int out_size, void* d_ws, size_t ws_size,
                              hipStream_t stream) {
    (void)in_sizes; (void)n_in; (void)out_size; (void)ws_size;
    const float* hidden   = (const float*)d_in[0];
    const float* w_router = (const float*)d_in[1];
    const float* wq = (const float*)d_in[2];
    const float* bq = (const float*)d_in[3];
    const float* wk = (const float*)d_in[4];
    const float* bk = (const float*)d_in[5];
    const float* wv = (const float*)d_in[6];
    const float* bv = (const float*)d_in[7];
    const float* wo = (const float*)d_in[8];
    const float* w_gate = (const float*)d_in[9];
    const float* w_up   = (const float*)d_in[10];
    const float* w_down = (const float*)d_in[11];
    const float* ln1 = (const float*)d_in[12];
    const float* ln2 = (const float*)d_in[13];
    float* out = (float*)d_out;

    char* ws = (char*)d_ws;
    size_t off = 0;
    auto take = [&](size_t b) -> char* {
        char* p = ws + off;
        off = (off + b + 255) & ~(size_t)255;
        return p;
    };
    float* scores = (float*)take(16384 * 4);
    int*   idx    = (int*)  take(2048 * 4);
    int*   pos    = (int*)  take(2048 * 4);
    float* gate   = (float*)take(2048 * 4);
    float* xsel   = (float*)take((size_t)2048 * 2048 * 4);
    unsigned short* hb   = (unsigned short*)take((size_t)2048 * 2048 * 2);
    float* qf   = (float*)take((size_t)2048 * 2048 * 4);   // pre-RoPE Q; reused as wo-out
    float* kf   = (float*)take((size_t)2048 * 1024 * 4);
    unsigned short* qbuf = (unsigned short*)take((size_t)2048 * 2048 * 2);
    unsigned short* kbuf = (unsigned short*)take((size_t)2048 * 1024 * 2);
    unsigned short* vbuf = (unsigned short*)take((size_t)2048 * 1024 * 2);
    unsigned short* obuf = (unsigned short*)take((size_t)2048 * 2048 * 2);
    float* x1   = (float*)take((size_t)2048 * 2048 * 4);
    unsigned short* h2b  = (unsigned short*)take((size_t)2048 * 2048 * 2);
    float* gbuf = (float*)take((size_t)2048 * 8192 * 4);
    unsigned short* actb = (unsigned short*)take((size_t)2048 * 8192 * 2);

    (void)hipMemcpyAsync(out, hidden, (size_t)4 * 4096 * 2048 * sizeof(float),
                         hipMemcpyDeviceToDevice, stream);

    router_kernel<<<2048, 256, 0, stream>>>(hidden, w_router, scores);
    topk_kernel<<<4, 1024, 0, stream>>>(scores, idx);
    gather_ln1<<<2048, 256, 0, stream>>>(hidden, scores, idx, ln1, xsel, hb, gate, pos);

    gemm_wmma<0><<<dim3(16, 16), 256, 0, stream>>>(hb, wq, bq, qf, nullptr, nullptr,
        nullptr, nullptr, nullptr, nullptr, 2048, 2048, 2048);
    gemm_wmma<0><<<dim3(8, 16), 256, 0, stream>>>(hb, wk, bk, kf, nullptr, nullptr,
        nullptr, nullptr, nullptr, nullptr, 2048, 1024, 2048);
    gemm_wmma<1><<<dim3(8, 16), 256, 0, stream>>>(hb, wv, bv, nullptr, vbuf, nullptr,
        nullptr, nullptr, nullptr, nullptr, 2048, 1024, 2048);

    rope_kernel<<<2048, 256, 0, stream>>>(qf, kf, pos, qbuf, kbuf);

    size_t flash_lds = (size_t)4 * 128 * QSTR * sizeof(unsigned short);
    flash_attn<<<dim3(16, 16), 256, flash_lds, stream>>>(qbuf, kbuf, vbuf, obuf);

    gemm_wmma<0><<<dim3(16, 16), 256, 0, stream>>>(obuf, wo, nullptr, qf, nullptr, nullptr,
        nullptr, nullptr, nullptr, nullptr, 2048, 2048, 2048);
    resid_ln2<<<2048, 256, 0, stream>>>(xsel, qf, ln2, x1, h2b);

    gemm_wmma<0><<<dim3(64, 16), 256, 0, stream>>>(h2b, w_gate, nullptr, gbuf, nullptr,
        nullptr, nullptr, nullptr, nullptr, nullptr, 2048, 8192, 2048);
    gemm_wmma<3><<<dim3(64, 16), 256, 0, stream>>>(h2b, w_up, nullptr, nullptr, actb,
        gbuf, nullptr, nullptr, nullptr, nullptr, 2048, 8192, 2048);
    gemm_wmma<4><<<dim3(16, 16), 256, 0, stream>>>(actb, w_down, nullptr, nullptr, nullptr,
        x1, xsel, gate, pos, out, 2048, 2048, 8192);
}
